// SingleModalAtten_57002805953290
// MI455X (gfx1250) — compile-verified, matrix operand order checked by reading, not run on hardware
//
#include <hip/hip_runtime.h>

// Problem constants (from reference): B=4, C=512, T=2048, 8 heads (ch=64), 32 groups.
#define B_   4
#define C_   512
#define T_   2048
#define NH_  8
#define BH_  32
#define CH_  64
#define NG_  32
#define CPG_ 16

#define DEV __device__ __forceinline__

typedef __attribute__((ext_vector_type(16))) __bf16 v16bf;
typedef __attribute__((ext_vector_type(8)))  float  v8f;
typedef __attribute__((ext_vector_type(4)))  unsigned u32x4;
typedef __attribute__((ext_vector_type(8)))  unsigned u32x8;

struct Frag { union { unsigned u[8]; v16bf v; }; };

DEV unsigned short f2bf(float f) {                // fp32 -> bf16 RNE
    unsigned u = __float_as_uint(f);
    return (unsigned short)((u + (0x7fffu + ((u >> 16) & 1u))) >> 16);
}
DEV unsigned pack2(float a, float b) {
    return (unsigned)f2bf(a) | ((unsigned)f2bf(b) << 16);
}
// K-pair base for 16-bit A/B fragment VGPR j (ISA 7.12.2: 16-bit A 16x32 layout)
DEV int kb(int j, int half) { return ((j < 4) ? 2 * j : 16 + 2 * (j - 4)) + 8 * half; }

// A-operand (or B-operand with n-major source): element (m,k) = base[m*ld + k]; pairs contiguous in k
DEV void load_frag_rowmajor(Frag& f, const unsigned short* base, int ld, int m, int half) {
    const unsigned short* row = base + (size_t)m * ld;
#pragma unroll
    for (int j = 0; j < 8; ++j) {
        int k = kb(j, half);
        f.u[j] = *(const unsigned*)(row + k);
    }
}
DEV v8f wmma_bf16(const Frag& a, const Frag& b, v8f c) {
    return __builtin_amdgcn_wmma_f32_16x16x32_bf16(false, a.v, false, b.v, (short)0, c, false, false);
}

DEV unsigned lds_of(const void* p) { return (unsigned)(size_t)p; }  // generic LDS ptr: low 32 bits = LDS addr

// ---- Tensor Data Mover: 2D bf16 tile load (global -> LDS), inline asm (CDNA5 ISA) ----
// D# group0: [1:0]=count=1 | lds_addr | global_addr[56:0] | type=2
// D# group1: data_size=1(2B) | tensor_dim0/1 | tile_dim0/1 (tile_dim2=0) | tensor_dim0_stride
DEV void tdm_load_2d(unsigned lds_addr, const unsigned short* gptr,
                     unsigned tensor_d0, unsigned tensor_d1,
                     unsigned tile_d0, unsigned tile_d1, unsigned stride_e) {
    unsigned long long ga = (unsigned long long)(size_t)gptr;
    u32x4 g0;
    g0[0] = 1u;                                                    // count=1 (valid user D#)
    g0[1] = lds_addr;
    g0[2] = (unsigned)ga;
    g0[3] = (unsigned)((ga >> 32) & 0x01FFFFFFull) | (2u << 30);   // addr[56:32] | type=2
    u32x8 g1;
    g1[0] = 1u << 16;                                              // data_size = 2 bytes
    g1[1] = (tensor_d0 & 0xffffu) << 16;                           // tensor_dim0[15:0] @ bits 63:48
    g1[2] = (tensor_d0 >> 16) | ((tensor_d1 & 0xffffu) << 16);     // td0[31:16] | td1[15:0]
    g1[3] = (tensor_d1 >> 16) | (tile_d0 << 16);                   // td1[31:16] | tile_dim0
    g1[4] = tile_d1;                                               // tile_dim1 | tile_dim2=0
    g1[5] = stride_e;                                              // tensor_dim0_stride[31:0]
    g1[6] = 0u;                                                    // stride[47:32] | dim1_stride lo
    g1[7] = 0u;
    asm volatile("tensor_load_to_lds %0, %1" :: "s"(g0), "s"(g1) : "memory");
}

// ---------------- Kernel 1: GroupNorm(32) -> bf16 hx, written T-MAJOR [b][t][C] ----------------
__global__ void gn_kernel(const float* __restrict__ x, const float* __restrict__ gs,
                          const float* __restrict__ gb, unsigned short* __restrict__ hxT) {
    int bg = blockIdx.x, b = bg / NG_, g = bg % NG_;
    const float* xp = x + ((size_t)(b * C_ + g * CPG_)) * T_;
    const int NEL = CPG_ * T_;
    float s = 0.f, ss = 0.f;
    for (int i = threadIdx.x; i < NEL; i += blockDim.x) { float v = xp[i]; s += v; ss += v * v; }
    __shared__ float r1[256], r2[256];
    r1[threadIdx.x] = s; r2[threadIdx.x] = ss; __syncthreads();
    for (int st = 128; st > 0; st >>= 1) {
        if ((int)threadIdx.x < st) { r1[threadIdx.x] += r1[threadIdx.x + st]; r2[threadIdx.x] += r2[threadIdx.x + st]; }
        __syncthreads();
    }
    float mean = r1[0] / (float)NEL;
    float var  = r2[0] / (float)NEL - mean * mean;
    float inv  = rsqrtf(var + 1e-5f);
    for (int i = threadIdx.x; i < NEL; i += blockDim.x) {
        int c = g * CPG_ + i / T_, t = i % T_;
        hxT[((size_t)b * T_ + t) * C_ + c] = f2bf((xp[i] - mean) * inv * gs[c] + gb[c]);
    }
}

// ---------------- Kernel 2: fp32 -> bf16 weight convert ----------------
__global__ void cvt_kernel(const float* __restrict__ w, unsigned short* __restrict__ o, int n) {
    int i = blockIdx.x * blockDim.x + threadIdx.x;
    if (i < n) o[i] = f2bf(w[i]);
}

// ---------------- Kernel 3: QKV GEMM (M=1536, N=2048/batch, K=512) ----------------
__global__ void __launch_bounds__(128) qkv_kernel(
        const unsigned short* __restrict__ wq, const unsigned short* __restrict__ hxT,
        const float* __restrict__ bq,
        unsigned short* __restrict__ qd, unsigned short* __restrict__ kd,
        unsigned short* __restrict__ vd) {
    int wv = blockIdx.x * (blockDim.x / 32) + (threadIdx.x >> 5);
    int lane = threadIdx.x & 31, n = lane & 15, half = lane >> 4;
    int b = wv / 1536, r = wv % 1536;
    int o0 = (r / 64) * 64, t0 = (r % 64) * 32;
    v8f acc[4][2];
#pragma unroll
    for (int mt = 0; mt < 4; ++mt)
#pragma unroll
        for (int nt = 0; nt < 2; ++nt) acc[mt][nt] = (v8f){};
    for (int k0 = 0; k0 < C_; k0 += 32) {
        Frag bf[2];
#pragma unroll
        for (int nt = 0; nt < 2; ++nt)   // B: (k=c, n=t) from t-major hx -> contiguous pairs
            load_frag_rowmajor(bf[nt], hxT + ((size_t)b * T_ + t0 + nt * 16) * C_ + k0, C_, n, half);
#pragma unroll
        for (int mt = 0; mt < 4; ++mt) {
            Frag af;
            load_frag_rowmajor(af, wq + (size_t)(o0 + mt * 16) * C_ + k0, C_, n, half);
#pragma unroll
            for (int nt = 0; nt < 2; ++nt) acc[mt][nt] = wmma_bf16(af, bf[nt], acc[mt][nt]);
        }
    }
    const float QKS = 0.35355339059f;  // 64^(-1/4), folded into both Q and K
#pragma unroll
    for (int mt = 0; mt < 4; ++mt) {
        int ob = o0 + mt * 16;
#pragma unroll
        for (int nt = 0; nt < 2; ++nt) {
            int t = t0 + nt * 16 + n;
            float v[8];
#pragma unroll
            for (int r8 = 0; r8 < 8; ++r8) v[r8] = acc[mt][nt][r8] + bq[ob + r8 + 8 * half];
            if (ob < 1024) {                       // Q or K: t-major [bh][t][64], 16B store
                int oo = (ob < 512) ? ob : ob - 512;
                unsigned short* dst = (ob < 512) ? qd : kd;
                int h = oo >> 6, cb = (oo & 63) + 8 * half;
                u32x4 pk;
#pragma unroll
                for (int j = 0; j < 4; ++j) pk[j] = pack2(v[2 * j] * QKS, v[2 * j + 1] * QKS);
                *(u32x4*)(dst + ((size_t)(b * NH_ + h) * T_ + t) * CH_ + cb) = pk;
            } else {                               // V: c-major [bh][64][T] (A-operand of PV wmma)
                int oo = ob - 1024, h = oo >> 6;
#pragma unroll
                for (int r8 = 0; r8 < 8; ++r8) {
                    int c = (oo & 63) + r8 + 8 * half;
                    vd[((size_t)(b * NH_ + h) * CH_ + c) * T_ + t] = f2bf(v[r8]);
                }
            }
        }
    }
}

// ---------------- Kernel 4: flash attention; 4 waves/WG share one head's K/V via TDM->LDS ----------------
__global__ void __launch_bounds__(128) att_kernel(
        const unsigned short* __restrict__ qd, const unsigned short* __restrict__ kd,
        const unsigned short* __restrict__ vd, unsigned short* __restrict__ aT) {
    __shared__ __align__(16) unsigned short ldsK[2][32 * CH_];  // [s(32)][c(64)]
    __shared__ __align__(16) unsigned short ldsV[2][CH_ * 32];  // [c(64)][s(32)]
    int wid = threadIdx.x >> 5, lane = threadIdx.x & 31, n = lane & 15, half = lane >> 4;
    int bh = blockIdx.x >> 4;                    // 32 heads, 16 blocks/head
    int t0 = (blockIdx.x & 15) * 128 + wid * 32; // 32 queries per wave
    const unsigned short* qh = qd + (size_t)bh * T_ * CH_;
    const unsigned short* kh = kd + (size_t)bh * T_ * CH_;
    const unsigned short* vh = vd + (size_t)bh * CH_ * T_;

    Frag qf[2][2];  // B-operand of S^T = K x Q : (k=c, n=t) from t-major Q
#pragma unroll
    for (int q = 0; q < 2; ++q)
#pragma unroll
        for (int ks = 0; ks < 2; ++ks)
            load_frag_rowmajor(qf[q][ks], qh + (size_t)(t0 + q * 16) * CH_ + ks * 32, CH_, n, half);

    v8f oacc[2][4];
#pragma unroll
    for (int q = 0; q < 2; ++q)
#pragma unroll
        for (int ct = 0; ct < 4; ++ct) oacc[q][ct] = (v8f){};
    float M[2] = {-3e38f, -3e38f}, L[2] = {0.f, 0.f};

    if (wid == 0) {  // first chunk: K tile [32s x 64c], V tile [64c x 32s]
        tdm_load_2d(lds_of(&ldsK[0][0]), kh, CH_, T_, CH_, 32, CH_);
        tdm_load_2d(lds_of(&ldsV[0][0]), vh, T_, CH_, 32, CH_, T_);
    }
    int cur = 0;
    for (int s0 = 0; s0 < T_; s0 += 32) {
        if (wid == 0) __builtin_amdgcn_s_wait_tensorcnt(0);
        __syncthreads();                          // publish buf[cur]; prev reads of buf[cur^1] done
        if (wid == 0 && s0 + 32 < T_) {           // prefetch next chunk while computing
            tdm_load_2d(lds_of(&ldsK[cur ^ 1][0]), kh + (size_t)(s0 + 32) * CH_, CH_, T_, CH_, 32, CH_);
            tdm_load_2d(lds_of(&ldsV[cur ^ 1][0]), vh + (s0 + 32), T_, CH_, 32, CH_, T_);
        }
        const unsigned short* Kl = &ldsK[cur][0];
        const unsigned short* Vl = &ldsV[cur][0];
        Frag kf[2][2], vf[4];
#pragma unroll
        for (int st = 0; st < 2; ++st)
#pragma unroll
            for (int ks = 0; ks < 2; ++ks)        // A of S^T: (m=s, k=c)
                load_frag_rowmajor(kf[st][ks], Kl + (st * 16) * CH_ + ks * 32, CH_, n, half);
#pragma unroll
        for (int ct = 0; ct < 4; ++ct)            // A of O^T = V^T x P: (m=c, k=s)
            load_frag_rowmajor(vf[ct], Vl + (ct * 16) * 32, 32, n, half);

#pragma unroll
        for (int q = 0; q < 2; ++q) {
            v8f sA = (v8f){}, sB = (v8f){};       // S^T tiles: m=s (two 16-row subtiles), n=t
            sA = wmma_bf16(kf[0][0], qf[q][0], sA);
            sA = wmma_bf16(kf[0][1], qf[q][1], sA);
            sB = wmma_bf16(kf[1][0], qf[q][0], sB);
            sB = wmma_bf16(kf[1][1], qf[q][1], sB);
            // online softmax over s (= m dim: in-lane regs + lane<->lane+16 swap)
            float mx = sA[0];
#pragma unroll
            for (int r8 = 0; r8 < 8; ++r8) { mx = fmaxf(mx, sA[r8]); mx = fmaxf(mx, sB[r8]); }
            mx = fmaxf(mx, __shfl_xor(mx, 16, 32));
            float Mn = fmaxf(M[q], mx);
            float alpha = __expf(M[q] - Mn);
            M[q] = Mn;
            float p0[8], p1[8], sum = 0.f;
#pragma unroll
            for (int r8 = 0; r8 < 8; ++r8) {
                p0[r8] = __expf(sA[r8] - Mn);
                p1[r8] = __expf(sB[r8] - Mn);
                sum += p0[r8] + p1[r8];
            }
            sum += __shfl_xor(sum, 16, 32);
            L[q] = L[q] * alpha + sum;
            // P(f32, D-layout) packs directly into the bf16 B-operand layout (k=s, n=t)
            Frag pf;
#pragma unroll
            for (int j = 0; j < 4; ++j) {
                pf.u[j]     = pack2(p0[2 * j], p0[2 * j + 1]);
                pf.u[4 + j] = pack2(p1[2 * j], p1[2 * j + 1]);
            }
#pragma unroll
            for (int ct = 0; ct < 4; ++ct) {
                v8f o = oacc[q][ct];
#pragma unroll
                for (int r8 = 0; r8 < 8; ++r8) o[r8] *= alpha;
                oacc[q][ct] = wmma_bf16(vf[ct], pf, o);
            }
        }
        cur ^= 1;
    }
    // epilogue: O^T (m=c, n=t) / L -> a stored T-MAJOR [b][t][C], 16B stores
    int b = bh >> 3, hc0 = (bh & 7) * CH_;
#pragma unroll
    for (int q = 0; q < 2; ++q) {
        float invL = 1.0f / L[q];
        int t = t0 + q * 16 + n;
#pragma unroll
        for (int ct = 0; ct < 4; ++ct) {
            u32x4 pk;
#pragma unroll
            for (int j = 0; j < 4; ++j)
                pk[j] = pack2(oacc[q][ct][2 * j] * invL, oacc[q][ct][2 * j + 1] * invL);
            *(u32x4*)(aT + ((size_t)b * T_ + t) * C_ + hc0 + ct * 16 + 8 * half) = pk;
        }
    }
}

// ---------------- Kernel 5: proj GEMM + bias + residual (fp32 out) ----------------
__global__ void __launch_bounds__(128) proj_kernel(
        const unsigned short* __restrict__ wp, const unsigned short* __restrict__ aT,
        const float* __restrict__ bp, const float* __restrict__ x, float* __restrict__ out) {
    int wv = blockIdx.x * (blockDim.x / 32) + (threadIdx.x >> 5);
    int lane = threadIdx.x & 31, n = lane & 15, half = lane >> 4;
    int b = wv / 512, r = wv % 512;
    int o0 = (r / 64) * 64, t0 = (r % 64) * 32;
    v8f acc[4][2];
#pragma unroll
    for (int mt = 0; mt < 4; ++mt)
#pragma unroll
        for (int nt = 0; nt < 2; ++nt) acc[mt][nt] = (v8f){};
    for (int k0 = 0; k0 < C_; k0 += 32) {
        Frag bf[2];
#pragma unroll
        for (int nt = 0; nt < 2; ++nt)   // B: (k=c, n=t) from t-major a -> contiguous pairs
            load_frag_rowmajor(bf[nt], aT + ((size_t)b * T_ + t0 + nt * 16) * C_ + k0, C_, n, half);
#pragma unroll
        for (int mt = 0; mt < 4; ++mt) {
            Frag af;
            load_frag_rowmajor(af, wp + (size_t)(o0 + mt * 16) * C_ + k0, C_, n, half);
#pragma unroll
            for (int nt = 0; nt < 2; ++nt) acc[mt][nt] = wmma_bf16(af, bf[nt], acc[mt][nt]);
        }
    }
#pragma unroll
    for (int mt = 0; mt < 4; ++mt)
#pragma unroll
        for (int nt = 0; nt < 2; ++nt) {
            int t = t0 + nt * 16 + n;
#pragma unroll
            for (int r8 = 0; r8 < 8; ++r8) {
                int o = o0 + mt * 16 + r8 + 8 * half;
                size_t idx = ((size_t)(b * C_ + o)) * T_ + t;
                out[idx] = x[idx] + acc[mt][nt][r8] + bp[o];
            }
        }
}

// ---------------- launch ----------------
extern "C" void kernel_launch(void* const* d_in, const int* in_sizes, int n_in,
                              void* d_out, int out_size, void* d_ws, size_t ws_size,
                              hipStream_t stream) {
    const float* x   = (const float*)d_in[0];
    const float* gs  = (const float*)d_in[1];
    const float* gb  = (const float*)d_in[2];
    const float* wq  = (const float*)d_in[3];
    const float* bq  = (const float*)d_in[4];
    const float* wp  = (const float*)d_in[5];
    const float* bp  = (const float*)d_in[6];
    float* out = (float*)d_out;

    // workspace carve-up (bf16 = ushort), 256B aligned; total ~44 MB
    unsigned char* ws = (unsigned char*)d_ws;
    size_t off = 0;
    auto carve = [&](size_t elems) {
        unsigned short* p = (unsigned short*)(ws + off);
        off += (elems * 2 + 255) & ~(size_t)255;
        return p;
    };
    const size_t NELT = (size_t)B_ * C_ * T_;              // 4,194,304
    unsigned short* hxT  = carve(NELT);                    // t-major [b][t][C]
    unsigned short* wqbf = carve((size_t)3 * C_ * C_);
    unsigned short* wpbf = carve((size_t)C_ * C_);
    unsigned short* qd   = carve(NELT);                    // t-major [bh][t][64]
    unsigned short* kd   = carve(NELT);                    // t-major [bh][t][64]
    unsigned short* vd   = carve(NELT);                    // c-major [bh][64][T]
    unsigned short* aT   = carve(NELT);                    // t-major [b][t][C]

    gn_kernel<<<B_ * NG_, 256, 0, stream>>>(x, gs, gb, hxT);
    cvt_kernel<<<(3 * C_ * C_ + 255) / 256, 256, 0, stream>>>(wq, wqbf, 3 * C_ * C_);
    cvt_kernel<<<(C_ * C_ + 255) / 256, 256, 0, stream>>>(wp, wpbf, C_ * C_);
    qkv_kernel<<<1536, 128, 0, stream>>>(wqbf, hxT, bq, qd, kd, vd);
    att_kernel<<<512, 128, 0, stream>>>(qd, kd, vd, aT);
    proj_kernel<<<512, 128, 0, stream>>>(wpbf, aT, bp, x, out);
}
